// GraphAttentionLayer_56822417326665
// MI455X (gfx1250) — matrix operand order repaired and checked by reading, not compiled, not run on hardware
//
#include <hip/hip_runtime.h>
#include <hip/hip_bf16.h>
#include <math.h>

// ---------------------------------------------------------------------------
// GAT layer for MI455X (gfx1250, wave32).
//   N=50000 nodes, F_IN=128, H=4 heads, C=32 -> F_OUT=128, E=800000 edges.
// Dominant cost: edge scatter/gather (~1GB, L2-resident on 192MB L2).
// GEMM uses v_wmma_f32_16x16x32_f16; A and B^T staged in LDS with 272B row
// stride so every per-lane fragment is two aligned 16B runs (b128 ds loads,
// bank-conflict-free across each half-wave).
// ---------------------------------------------------------------------------

#define N_NODES 50000
#define F_INDIM 128
#define F_OUT   128
#define NHEADS  4
#define CDIM    32
#define NEDGES  800000
#define NMSG    (NEDGES + N_NODES)   // edges + self loops
#define NEG_SLOPE 0.2f

typedef __attribute__((ext_vector_type(16))) _Float16 v16h;
typedef __attribute__((ext_vector_type(8)))  _Float16 v8h;
typedef __attribute__((ext_vector_type(8)))  float    v8f;

#define AS_LD 136   // halves per LDS row = 272B: 16B-aligned runs, banks 4*lane
#define BS_LD 136

// -------------------- kernel 0: init out=bias, m=-inf, denom=0 -------------
__global__ __launch_bounds__(256) void gat_init(float* __restrict__ out,
                                                const float* __restrict__ bias,
                                                float* __restrict__ mbuf,
                                                float* __restrict__ denom) {
  int t = blockIdx.x * 256 + threadIdx.x;
  if (t < N_NODES * F_OUT) out[t] = bias[t & (F_OUT - 1)];
  if (t < N_NODES * NHEADS) { mbuf[t] = -INFINITY; denom[t] = 0.0f; }
}

// -------------------- kernel 1: h = x @ W via WMMA f16 ---------------------
// Block = 256 threads (8 waves). Block covers 16 rows x 128 cols.
// Wave w computes the 16x16 tile at cols [16w, 16w+16). K=128 in 4 chunks of 32.
__global__ __launch_bounds__(256) void gat_gemm(const float* __restrict__ x,
                                                const float* __restrict__ W,
                                                float* __restrict__ h) {
  __shared__ __align__(32) _Float16 As[16 * AS_LD];    // x tile, f16, [row][K]
  __shared__ __align__(32) _Float16 Bs[128 * BS_LD];   // W^T,    f16, [col][K]

  const int tid  = threadIdx.x;
  const int row0 = blockIdx.x * 16;

  // Cooperative loads (coalesced global reads), f32 -> f16 convert.
  // W stored transposed so per-lane B fragments are contiguous in K.
  for (int idx = tid; idx < 128 * 128; idx += 256) {
    int k = idx >> 7, n = idx & 127;
    Bs[n * BS_LD + k] = (_Float16)W[idx];
  }
  for (int idx = tid; idx < 16 * 128; idx += 256) {
    int m = idx >> 7, k = idx & 127;
    As[m * AS_LD + k] = (_Float16)x[(row0 + m) * F_INDIM + k];
  }
  __syncthreads();

  const int lane = tid & 31;
  const int wave = tid >> 5;
  const int col0 = wave * 16;
  const int mn   = lane & 15;      // row (A) / col (B,D) within tile
  const int g    = lane >> 4;      // half-wave group

  const _Float16* aRow = As + mn * AS_LD;
  const _Float16* bRow = Bs + (col0 + mn) * BS_LD;

  v8f c = {};
#pragma unroll
  for (int kk = 0; kk < 4; ++kk) {
    const int kbase = kk * 32;
    // A 16x32 f16 layout (ISA 7.12.2): lane(m,g) holds K = kbase+8g+[0,8)
    // and K = kbase+16+8g+[0,8)  -> two aligned 16B runs.
    v8h a0 = *(const v8h*)(aRow + kbase + g * 8);
    v8h a1 = *(const v8h*)(aRow + kbase + 16 + g * 8);
    // B 32x16 f16 layout: lane(n,g) holds K = kbase+8g+[0,8) and
    // K = kbase+16+8g+[0,8) of column n -> two aligned 16B runs in the transposed store.
    v8h b0 = *(const v8h*)(bRow + kbase + g * 8);
    v8h b1 = *(const v8h*)(bRow + kbase + 16 + g * 8);
    v16h a = __builtin_shufflevector(a0, a1, 0, 1, 2, 3, 4, 5, 6, 7,
                                     8, 9, 10, 11, 12, 13, 14, 15);
    v16h b = __builtin_shufflevector(b0, b1, 0, 1, 2, 3, 4, 5, 6, 7,
                                     8, 9, 10, 11, 12, 13, 14, 15);
    c = __builtin_amdgcn_wmma_f32_16x16x32_f16(
        /*neg_a=*/false, a, /*neg_b=*/false, b,
        /*c_mod=*/(short)0, c, /*reuse_a=*/false, /*reuse_b=*/false);
  }

  // D layout: VGPR r -> row M = r + 8*g, col = mn
#pragma unroll
  for (int r = 0; r < 8; ++r) {
    h[(row0 + r + 8 * g) * F_OUT + col0 + mn] = c[r];
  }
}

// -------------------- kernel 2: per-node logits ----------------------------
__global__ __launch_bounds__(256) void gat_logits(const float* __restrict__ h,
                                                  const float* __restrict__ att_src,
                                                  const float* __restrict__ att_dst,
                                                  float* __restrict__ a_src,
                                                  float* __restrict__ a_dst) {
  int t = blockIdx.x * 256 + threadIdx.x;
  if (t >= N_NODES * NHEADS) return;
  int n = t >> 2, hh = t & 3;
  const float* hp = h + n * F_OUT + hh * CDIM;
  const float* as = att_src + hh * CDIM;
  const float* ad = att_dst + hh * CDIM;
  float s = 0.0f, d = 0.0f;
#pragma unroll
  for (int cidx = 0; cidx < CDIM; ++cidx) {
    float v = hp[cidx];
    s += v * as[cidx];
    d += v * ad[cidx];
  }
  a_src[t] = s;
  a_dst[t] = d;
}

// -------------------- helpers ----------------------------------------------
__device__ __forceinline__ void atomicMaxFloat(float* addr, float v) {
  // Sign-aware IEEE bit trick: positives via signed max, negatives via unsigned min.
  if (v >= 0.0f) atomicMax((int*)addr, __float_as_int(v));
  else           atomicMin((unsigned int*)addr, __float_as_uint(v));
}

__device__ __forceinline__ void loadEdge(const long long* __restrict__ ei,
                                         int t, int* s, int* d) {
  if (t < NEDGES) { *s = (int)ei[t]; *d = (int)ei[NEDGES + t]; }
  else            { *s = *d = t - NEDGES; }          // self loop
}

__device__ __forceinline__ float lrelu(float x) {
  return x > 0.0f ? x : NEG_SLOPE * x;
}

// -------------------- kernel 3: segment max over incoming edges ------------
__global__ __launch_bounds__(256) void gat_edge_max(const long long* __restrict__ ei,
                                                    const float* __restrict__ a_src,
                                                    const float* __restrict__ a_dst,
                                                    float* __restrict__ mbuf) {
  int t = blockIdx.x * 256 + threadIdx.x;
  if (t >= NMSG) return;
  int s, d;
  loadEdge(ei, t, &s, &d);
  const float4 as = *(const float4*)(a_src + s * 4);
  const float4 ad = *(const float4*)(a_dst + d * 4);
  atomicMaxFloat(mbuf + d * 4 + 0, lrelu(as.x + ad.x));
  atomicMaxFloat(mbuf + d * 4 + 1, lrelu(as.y + ad.y));
  atomicMaxFloat(mbuf + d * 4 + 2, lrelu(as.z + ad.z));
  atomicMaxFloat(mbuf + d * 4 + 3, lrelu(as.w + ad.w));
}

// -------------------- kernel 4: exp + segment sum, cache e_exp -------------
__global__ __launch_bounds__(256) void gat_edge_exp(const long long* __restrict__ ei,
                                                    const float* __restrict__ a_src,
                                                    const float* __restrict__ a_dst,
                                                    const float* __restrict__ mbuf,
                                                    float* __restrict__ e_exp,
                                                    float* __restrict__ denom) {
  int t = blockIdx.x * 256 + threadIdx.x;
  if (t >= NMSG) return;
  int s, d;
  loadEdge(ei, t, &s, &d);
  const float4 as = *(const float4*)(a_src + s * 4);
  const float4 ad = *(const float4*)(a_dst + d * 4);
  const float4 mm = *(const float4*)(mbuf + d * 4);
  float4 w;
  w.x = __expf(lrelu(as.x + ad.x) - mm.x);
  w.y = __expf(lrelu(as.y + ad.y) - mm.y);
  w.z = __expf(lrelu(as.z + ad.z) - mm.z);
  w.w = __expf(lrelu(as.w + ad.w) - mm.w);
  *(float4*)(e_exp + (size_t)t * 4) = w;
  unsafeAtomicAdd(denom + d * 4 + 0, w.x);
  unsafeAtomicAdd(denom + d * 4 + 1, w.y);
  unsafeAtomicAdd(denom + d * 4 + 2, w.z);
  unsafeAtomicAdd(denom + d * 4 + 3, w.w);
}

// -------------------- kernel 5: weighted scatter-add (one wave per edge) ---
// Lane l handles channels [4l, 4l+4): 512B fully-coalesced gather of h[src],
// 4 f32 atomics into out[dst]. head = l/8 (8 lanes per 32-wide head).
__global__ __launch_bounds__(256) void gat_scatter(const long long* __restrict__ ei,
                                                   const float* __restrict__ h,
                                                   const float* __restrict__ e_exp,
                                                   const float* __restrict__ denom,
                                                   float* __restrict__ out) {
  int wid  = blockIdx.x * 8 + (threadIdx.x >> 5);  // global wave id == message id
  int lane = threadIdx.x & 31;
  if (wid >= NMSG) return;
  int s, d;
  loadEdge(ei, wid, &s, &d);
  const int head = lane >> 3;
  const float att = e_exp[(size_t)wid * 4 + head] / denom[d * 4 + head];
  const float4 hv = *(const float4*)(h + (size_t)s * F_OUT + lane * 4);
  float* op = out + (size_t)d * F_OUT + lane * 4;
  unsafeAtomicAdd(op + 0, hv.x * att);
  unsafeAtomicAdd(op + 1, hv.y * att);
  unsafeAtomicAdd(op + 2, hv.z * att);
  unsafeAtomicAdd(op + 3, hv.w * att);
}

// ---------------------------------------------------------------------------
extern "C" void kernel_launch(void* const* d_in, const int* in_sizes, int n_in,
                              void* d_out, int out_size, void* d_ws, size_t ws_size,
                              hipStream_t stream) {
  const float*     x        = (const float*)d_in[0];
  const long long* edge_idx = (const long long*)d_in[1];   // int64 [2, E]
  const float*     W        = (const float*)d_in[2];
  const float*     att_src  = (const float*)d_in[3];
  const float*     att_dst  = (const float*)d_in[4];
  const float*     bias     = (const float*)d_in[5];
  float*           out      = (float*)d_out;

  // Workspace layout (all 256B-aligned by construction)
  char* ws = (char*)d_ws;
  float* h      = (float*)ws;                 ws += (size_t)N_NODES * F_OUT * 4;   // 25.6 MB
  float* a_src  = (float*)ws;                 ws += (size_t)N_NODES * NHEADS * 4;  // 0.8 MB
  float* a_dst  = (float*)ws;                 ws += (size_t)N_NODES * NHEADS * 4;
  float* mbuf   = (float*)ws;                 ws += (size_t)N_NODES * NHEADS * 4;
  float* denom  = (float*)ws;                 ws += (size_t)N_NODES * NHEADS * 4;
  float* e_exp  = (float*)ws;                 ws += (size_t)NMSG * NHEADS * 4;     // 13.6 MB

  // 0) init: out = bias broadcast, m = -inf, denom = 0
  gat_init<<<(N_NODES * F_OUT + 255) / 256, 256, 0, stream>>>(out, bias, mbuf, denom);

  // 1) h = x @ W  (WMMA f16, f32 accumulate)
  gat_gemm<<<N_NODES / 16, 256, 0, stream>>>(x, W, h);

  // 2) per-node logits
  gat_logits<<<(N_NODES * NHEADS + 255) / 256, 256, 0, stream>>>(h, att_src, att_dst,
                                                                 a_src, a_dst);

  // 3) segment max
  gat_edge_max<<<(NMSG + 255) / 256, 256, 0, stream>>>(edge_idx, a_src, a_dst, mbuf);

  // 4) exp + segment sum
  gat_edge_exp<<<(NMSG + 255) / 256, 256, 0, stream>>>(edge_idx, a_src, a_dst, mbuf,
                                                       e_exp, denom);

  // 5) weighted scatter-add, one wave per message
  gat_scatter<<<(NMSG + 7) / 8, 256, 0, stream>>>(edge_idx, h, e_exp, denom, out);
}
